// MultiScaleGambaLayer_7653631722028
// MI455X (gfx1250) — matrix operand order repaired and checked by reading, not compiled, not run on hardware
//
#include <hip/hip_runtime.h>

// ---------------------------------------------------------------------------
// MultiScaleGamba layer, MI455X (gfx1250) implementation.
//
// Input order assumption (harness flattens setup_inputs() dict; `params`
// flattened as a jax pytree => dict keys sorted alphabetically per level):
//   0 x[N*64] f32, 1 edge_index[2*E] i32, 2 edge_attr[E*2] f32, 3 batch[N] i32
//   per conv c (base 4+12c): bW,bb, mlp{W1,W2,b1,b2,bt,g}, nb,ng, sW,sb
//   40 ln_b, 41 ln_g
//   mamba (base 42): A_log,D,conv_W,conv_b,dt_W,dt_b,fnorm_w,in_W,norm_w,out_W,xproj_W
//   merge (base 53): W1,W2,b1,b2,bt,g
//   59 theta
// ---------------------------------------------------------------------------

#define HID   64
#define NB_G  8          // graphs
#define KTOK  32
#define IDIM  128
#define SSZ   16
#define EPSLN 1e-5f

typedef __attribute__((ext_vector_type(16))) _Float16 v16h;
typedef __attribute__((ext_vector_type(8)))  float    v8f;

union Frag16 { v16h v; uint4 u[2]; };

__device__ __forceinline__ v8f wmma16(v16h a, v16h b, v8f c) {
  return __builtin_amdgcn_wmma_f32_16x16x32_f16(false, a, false, b, (short)0, c,
                                                false, false);
}
// DS ops are in-order within a wave; this only needs to stop the compiler from
// reordering LDS traffic across the fence (and drains DScnt).
__device__ __forceinline__ void lds_fence() {
  asm volatile("s_wait_dscnt 0" ::: "memory");
}

struct EdgeP { const float* p[3][8]; };  // [conv][b1,b2,g,bt,sW,sb,bW,bb]
struct NodeP { const float* ng[3]; const float* nb[3]; };
struct MamP  { const float *A_log,*D,*conv_W,*conv_b,*dt_W,*dt_b,*fnorm_w,
                           *in_W,*norm_w,*out_W,*xproj_W; };

// --------------------------- small utility kernels -------------------------

__global__ void k_cvt_f16(const float* __restrict__ s, _Float16* __restrict__ d, int n) {
  for (int i = blockIdx.x * blockDim.x + threadIdx.x; i < n; i += gridDim.x * blockDim.x)
    d[i] = (_Float16)s[i];
}

__global__ void k_zero_f32(float* __restrict__ p, int n) {
  for (int i = blockIdx.x * blockDim.x + threadIdx.x; i < n; i += gridDim.x * blockDim.x)
    p[i] = 0.f;
}

// Re-pack a row-major f32 weight [K x 64] into WMMA B-fragment order:
// frag fi=kt*4+nt holds 32 lanes x 16 halfs (lane<16: K=kt*32+{0..7,16..23},
// lane>=16: K=kt*32+{8..15,24..31}; col = nt*16 + lane%16), lane-contiguous 32B.
__global__ void k_prep_frag(const float* __restrict__ src, _Float16* __restrict__ dst, int K) {
  int total = (K / 32) * 4 * 512;
  for (int idx = blockIdx.x * blockDim.x + threadIdx.x; idx < total;
       idx += gridDim.x * blockDim.x) {
    int fi   = idx >> 9;
    int lane = (idx >> 4) & 31;
    int i    = idx & 15;
    int kt = fi >> 2, nt = fi & 3;
    int col = nt * 16 + (lane & 15);
    int kk  = kt * 32 + ((lane >> 4) ? 8 : 0) + (i & 7) + ((i >> 3) << 4);
    dst[idx] = (_Float16)src[kk * HID + col];
  }
}

// ------------------------------ edge MLP (WMMA) ----------------------------
// One wave = 16 edges. Gather A once (shared by all 3 convs), then per conv:
// WMMA GEMM1(128->64) +b1 -> LDS -> LN(g,bt) -> ReLU -> f16 -> WMMA GEMM2
// (64->64) +b2 -> *sigmoid(edge gate) -> f32 atomic scatter into agg[c].
__global__ __launch_bounds__(128) void k_edge(
    const _Float16* __restrict__ xh, const int* __restrict__ ei,
    const float* __restrict__ ea, const _Float16* __restrict__ w1h,
    const _Float16* __restrict__ w2h, EdgeP P, float* __restrict__ agg,
    int ntiles, int E_, int N_) {
  __shared__ float    lt4[4][1024];
  __shared__ _Float16 lh4[4][1024];
  __shared__ float    sprm[3][8][HID];
  for (int idx = threadIdx.x; idx < 3 * 8 * HID; idx += blockDim.x) {
    int c = idx / 512, q = (idx >> 6) & 7, f = idx & 63;
    sprm[c][q][f] = P.p[c][q][f];
  }
  __syncthreads();

  const int lane = threadIdx.x & 31;
  const int wave = threadIdx.x >> 5;
  const int r  = lane & 15;
  const int hi = lane >> 4;
  float*    lt = lt4[wave];
  _Float16* lh = lh4[wave];
  const int* srcp = ei;
  const int* dstp = ei + E_;
  const v8f vzero = {};

  int wgl = blockIdx.x * 4 + wave;
  int nw  = gridDim.x * 4;
  for (int tile = wgl; tile < ntiles; tile += nw) {
    int   e  = tile * 16 + r;               // lanes L and L+16 share a row
    int   dn = dstp[e], sn = srcp[e];
    float e0 = ea[2 * e], e1 = ea[2 * e + 1];

    int rowd[8]; float re0[8], re1[8];      // per-C-row metadata (row = v+8*hi)
#pragma unroll
    for (int v = 0; v < 8; ++v) {
      int rr  = v + 8 * hi;
      rowd[v] = __shfl(dn, rr);
      re0[v]  = __shfl(e0, rr);
      re1[v]  = __shfl(e1, rr);
    }

    Frag16 a[4];                            // A = [16 x 128] = x[dst] || x[src]
#pragma unroll
    for (int kt = 0; kt < 4; ++kt) {
      int node = (kt < 2) ? dn : sn;
      int f0   = (kt & 1) * 32 + hi * 8;
      const uint4* p = (const uint4*)(xh + (size_t)node * HID + f0);
      a[kt].u[0] = p[0];
      a[kt].u[1] = p[2];                    // +16 halfs
    }

    for (int c = 0; c < 3; ++c) {
      const v16h* W1f = (const v16h*)(w1h + c * 16 * 512);
      const v16h* W2f = (const v16h*)(w2h + c * 8 * 512);
      v8f acc[4];
#pragma unroll
      for (int nt = 0; nt < 4; ++nt) acc[nt] = vzero;
#pragma unroll
      for (int kt = 0; kt < 4; ++kt)
#pragma unroll
        for (int nt = 0; nt < 4; ++nt)
          acc[nt] = wmma16(a[kt].v, W1f[(kt * 4 + nt) * 32 + lane], acc[nt]);

#pragma unroll
      for (int nt = 0; nt < 4; ++nt) {
        int colf  = nt * 16 + r;
        float b1v = sprm[c][0][colf];
#pragma unroll
        for (int v = 0; v < 8; ++v)
          lt[(v + 8 * hi) * HID + colf] = acc[nt][v] + b1v;
      }
      lds_fence();

      // LayerNorm + ReLU: lane handles (row=r, half=hi)
      float sum = 0.f, sq = 0.f;
#pragma unroll 8
      for (int j = 0; j < 32; ++j) {
        float t = lt[r * HID + hi * 32 + j];
        sum += t; sq += t * t;
      }
      sum += __shfl_xor(sum, 16);
      sq  += __shfl_xor(sq, 16);
      float mean = sum * (1.f / 64.f);
      float rs   = rsqrtf(sq * (1.f / 64.f) - mean * mean + EPSLN);
#pragma unroll 8
      for (int j = 0; j < 32; ++j) {
        int f   = hi * 32 + j;
        float t = (lt[r * HID + f] - mean) * rs * sprm[c][2][f] + sprm[c][3][f];
        lh[r * HID + f] = (_Float16)fmaxf(t, 0.f);
      }
      lds_fence();

      Frag16 a2[2];
#pragma unroll
      for (int k2 = 0; k2 < 2; ++k2) {
        int f0 = k2 * 32 + hi * 8;
        a2[k2].u[0] = *(const uint4*)&lh[r * HID + f0];
        a2[k2].u[1] = *(const uint4*)&lh[r * HID + f0 + 16];
      }
      v8f acc2[4];
#pragma unroll
      for (int nt = 0; nt < 4; ++nt) acc2[nt] = vzero;
#pragma unroll
      for (int k2 = 0; k2 < 2; ++k2)
#pragma unroll
        for (int nt = 0; nt < 4; ++nt)
          acc2[nt] = wmma16(a2[k2].v, W2f[(k2 * 4 + nt) * 32 + lane], acc2[nt]);

#pragma unroll
      for (int nt = 0; nt < 4; ++nt) {
        int colf  = nt * 16 + r;
        float b2v = sprm[c][1][colf];
        float sWv = sprm[c][4][colf], sbv = sprm[c][5][colf];
        float bWv = sprm[c][6][colf], bbv = sprm[c][7][colf];
#pragma unroll
        for (int v = 0; v < 8; ++v) {
          float t   = re0[v] * sWv + re1[v] * bWv + sbv + bbv;
          float wgt = 1.f / (1.f + __expf(-t));
          float val = (acc2[nt][v] + b2v) * wgt;
          unsafeAtomicAdd(&agg[(size_t)c * N_ * HID + (size_t)rowd[v] * HID + colf], val);
        }
      }
    }
  }
}

// ----------------------- node LN after aggregation -------------------------
// xcat[n, c*64+f] = f16( LN(x[n]+agg[c][n]; ng,nb) ), one wave per (c,n).
__global__ __launch_bounds__(256) void k_node_ln(
    const float* __restrict__ x, const float* __restrict__ agg,
    _Float16* __restrict__ xcat, NodeP P, int N_) {
  int lane = threadIdx.x & 31;
  int wid  = (blockIdx.x * blockDim.x + threadIdx.x) >> 5;
  int nw   = (gridDim.x * blockDim.x) >> 5;
  for (int row = wid; row < 3 * N_; row += nw) {
    int c = row / N_;
    int n = row - c * N_;
    const float* xr = x + (size_t)n * HID;
    const float* ar = agg + (size_t)c * N_ * HID + (size_t)n * HID;
    float v0 = xr[lane] + ar[lane];
    float v1 = xr[lane + 32] + ar[lane + 32];
    float sum = v0 + v1, sq = v0 * v0 + v1 * v1;
    for (int o = 1; o < 32; o <<= 1) { sum += __shfl_xor(sum, o); sq += __shfl_xor(sq, o); }
    float mean = sum * (1.f / 64.f);
    float rs   = rsqrtf(sq * (1.f / 64.f) - mean * mean + EPSLN);
    const float* ng = P.ng[c];
    const float* nb = P.nb[c];
    _Float16* dst = xcat + (size_t)n * 256 + c * HID;
    dst[lane]      = (_Float16)((v0 - mean) * rs * ng[lane] + nb[lane]);
    dst[lane + 32] = (_Float16)((v1 - mean) * rs * ng[lane + 32] + nb[lane + 32]);
  }
}

// --------------------- Gram matrices: G[b] = X_b^T X_b ---------------------
__global__ __launch_bounds__(256) void k_gram(const float* __restrict__ x,
                                              float* __restrict__ G, int Nb) {
  __shared__ float sA[128 * 16];
  __shared__ float sB[128 * 16];
  int b   = blockIdx.x >> 4;
  int t16 = blockIdx.x & 15;
  int hiB = (t16 >> 2) * 16, hjB = (t16 & 3) * 16;
  int hiL = threadIdx.x >> 4, hjL = threadIdx.x & 15;
  const float* xb = x + (size_t)b * Nb * HID;
  float acc = 0.f;
  for (int c0 = 0; c0 < Nb; c0 += 128) {
    __syncthreads();
    for (int idx = threadIdx.x; idx < 128 * 16; idx += 256) {
      int nl = idx >> 4, f = idx & 15;
      int n  = c0 + nl;
      float va = 0.f, vb = 0.f;
      if (n < Nb) {
        va = xb[(size_t)n * HID + hiB + f];
        vb = xb[(size_t)n * HID + hjB + f];
      }
      sA[idx] = va; sB[idx] = vb;
    }
    __syncthreads();
    int lim = Nb - c0; if (lim > 128) lim = 128;
    for (int nl = 0; nl < lim; ++nl)
      acc += sA[nl * 16 + hiL] * sB[nl * 16 + hjL];
  }
  G[((size_t)b * HID + hiB + hiL) * HID + hjB + hjL] = acc;
}

// alphaX[b] = theta^T @ G[b]     ([32 x 64])
__global__ void k_alphax(const float* __restrict__ theta, const float* __restrict__ G,
                         float* __restrict__ aX) {
  int b = blockIdx.x;
  for (int idx = threadIdx.x; idx < KTOK * HID; idx += blockDim.x) {
    int k = idx >> 6, ho = idx & 63;
    float acc = 0.f;
    for (int h = 0; h < HID; ++h)
      acc += theta[h * KTOK + k] * G[(size_t)b * HID * HID + h * HID + ho];
    aX[(size_t)b * KTOK * HID + idx] = acc;
  }
}

// ------------------------------- Mamba block -------------------------------
// One block per graph, 128 threads (thread = intermediate channel), all state
// in LDS. Only the last token (k=31) of the output is used downstream.
__global__ __launch_bounds__(128) void k_mamba(const float* __restrict__ aX, MamP P,
                                               _Float16* __restrict__ xmh) {
  __shared__ float s_h [KTOK * HID];
  __shared__ float s_hs[KTOK * HID];
  __shared__ float s_x [KTOK * IDIM];
  __shared__ float s_g [KTOK * IDIM];
  __shared__ float s_xc[KTOK * IDIM];
  __shared__ float s_sp[KTOK * 36];
  __shared__ float s_dt[KTOK * IDIM];
  __shared__ float s_y [IDIM];
  __shared__ float s_o [HID];
  __shared__ float s_rs;
  const int b = blockIdx.x, tid = threadIdx.x;

  for (int idx = tid; idx < KTOK * HID; idx += 128)
    s_h[idx] = aX[(size_t)b * KTOK * HID + idx];
  __syncthreads();
  if (tid < KTOK) {                                   // RMSNorm rows
    float sq = 0.f;
    for (int f = 0; f < HID; ++f) { float t = s_h[tid * HID + f]; sq += t * t; }
    float rs = rsqrtf(sq * (1.f / HID) + EPSLN);
    for (int f = 0; f < HID; ++f)
      s_hs[tid * HID + f] = s_h[tid * HID + f] * rs * P.norm_w[f];
  }
  __syncthreads();
  for (int k = 0; k < KTOK; ++k) {                    // in-proj 64 -> 256
    float a0 = 0.f, a1 = 0.f;
    for (int f = 0; f < HID; ++f) {
      float h = s_hs[k * HID + f];
      a0 += h * P.in_W[f * 256 + tid];
      a1 += h * P.in_W[f * 256 + 128 + tid];
    }
    s_x[k * IDIM + tid] = a0;
    s_g[k * IDIM + tid] = a1;
  }
  __syncthreads();
  {                                                   // causal conv(4) + silu
    float cw0 = P.conv_W[tid * 4], cw1 = P.conv_W[tid * 4 + 1];
    float cw2 = P.conv_W[tid * 4 + 2], cw3 = P.conv_W[tid * 4 + 3];
    float cb = P.conv_b[tid];
    for (int k = 0; k < KTOK; ++k) {
      float s = cb + cw3 * s_x[k * IDIM + tid];
      if (k >= 1) s += cw2 * s_x[(k - 1) * IDIM + tid];
      if (k >= 2) s += cw1 * s_x[(k - 2) * IDIM + tid];
      if (k >= 3) s += cw0 * s_x[(k - 3) * IDIM + tid];
      s_xc[k * IDIM + tid] = s / (1.f + __expf(-s));
    }
  }
  __syncthreads();
  for (int idx = tid; idx < KTOK * 36; idx += 128) {  // x-proj 128 -> 36
    int k = idx / 36, c = idx - k * 36;
    float acc = 0.f;
    for (int i = 0; i < IDIM; ++i) acc += s_xc[k * IDIM + i] * P.xproj_W[i * 36 + c];
    s_sp[idx] = acc;
  }
  __syncthreads();
  {                                                   // dt = softplus(dt_raw@dt_W + b)
    float w0 = P.dt_W[tid], w1 = P.dt_W[IDIM + tid];
    float w2 = P.dt_W[2 * IDIM + tid], w3 = P.dt_W[3 * IDIM + tid];
    float db = P.dt_b[tid];
    for (int k = 0; k < KTOK; ++k) {
      float t = db + s_sp[k * 36] * w0 + s_sp[k * 36 + 1] * w1 +
                s_sp[k * 36 + 2] * w2 + s_sp[k * 36 + 3] * w3;
      s_dt[k * IDIM + tid] = (t > 20.f) ? t : log1pf(__expf(t));
    }
  }
  __syncthreads();
  {                                                   // selective scan
    float as[SSZ], st[SSZ];
#pragma unroll
    for (int s = 0; s < SSZ; ++s) { as[s] = -__expf(P.A_log[tid * SSZ + s]); st[s] = 0.f; }
    float y31 = 0.f;
    for (int k = 0; k < KTOK; ++k) {
      float dtv = s_dt[k * IDIM + tid];
      float dx  = dtv * s_xc[k * IDIM + tid];
#pragma unroll
      for (int s = 0; s < SSZ; ++s)
        st[s] = __expf(dtv * as[s]) * st[s] + dx * s_sp[k * 36 + 4 + s];
      if (k == KTOK - 1) {
#pragma unroll
        for (int s = 0; s < SSZ; ++s) y31 += st[s] * s_sp[k * 36 + 20 + s];
      }
    }
    y31 += P.D[tid] * s_xc[(KTOK - 1) * IDIM + tid];
    float g = s_g[(KTOK - 1) * IDIM + tid];
    y31 *= g / (1.f + __expf(-g));
    s_y[tid] = y31;
  }
  __syncthreads();
  if (tid < HID) {                                    // out-proj + residual (row 31)
    float o = s_h[(KTOK - 1) * HID + tid];
    for (int i = 0; i < IDIM; ++i) o += s_y[i] * P.out_W[i * HID + tid];
    s_o[tid] = o;
  }
  __syncthreads();
  if (tid == 0) {
    float sq = 0.f;
    for (int f = 0; f < HID; ++f) sq += s_o[f] * s_o[f];
    s_rs = rsqrtf(sq * (1.f / HID) + EPSLN);
  }
  __syncthreads();
  if (tid < HID)
    xmh[b * HID + tid] = (_Float16)(s_o[tid] * s_rs * P.fnorm_w[tid]);
}

__global__ void k_fill_xm(const _Float16* __restrict__ xmh, const int* __restrict__ batch,
                          _Float16* __restrict__ xcat, int total) {
  for (int idx = blockIdx.x * blockDim.x + threadIdx.x; idx < total;
       idx += gridDim.x * blockDim.x) {
    int n = idx >> 6, f = idx & 63;
    xcat[(size_t)n * 256 + 192 + f] = xmh[batch[n] * HID + f];
  }
}

// ------------------------------ merge MLP (WMMA) ---------------------------
__global__ __launch_bounds__(128) void k_merge(
    const _Float16* __restrict__ xcat, const _Float16* __restrict__ mw1,
    const _Float16* __restrict__ mw2, const float* __restrict__ b1,
    const float* __restrict__ gg, const float* __restrict__ bt,
    const float* __restrict__ b2, const float* __restrict__ x,
    const float* __restrict__ lng, const float* __restrict__ lnb,
    float* __restrict__ out, int ntiles) {
  __shared__ float    lt4[4][1024];
  __shared__ _Float16 lh4[4][1024];
  const int lane = threadIdx.x & 31;
  const int wave = threadIdx.x >> 5;
  const int r = lane & 15, hi = lane >> 4;
  float*    lt = lt4[wave];
  _Float16* lh = lh4[wave];
  const v16h* W1f = (const v16h*)mw1;
  const v16h* W2f = (const v16h*)mw2;
  const v8f vzero = {};
  int wgl = blockIdx.x * 4 + wave;
  int nw  = gridDim.x * 4;
  for (int tile = wgl; tile < ntiles; tile += nw) {
    int node = tile * 16 + r;
    v8f acc[4];
#pragma unroll
    for (int nt = 0; nt < 4; ++nt) acc[nt] = vzero;
#pragma unroll
    for (int kt = 0; kt < 8; ++kt) {                  // K = 256
      Frag16 a;
      int f0 = kt * 32 + hi * 8;
      const uint4* p = (const uint4*)(xcat + (size_t)node * 256 + f0);
      a.u[0] = p[0]; a.u[1] = p[2];
#pragma unroll
      for (int nt = 0; nt < 4; ++nt)
        acc[nt] = wmma16(a.v, W1f[(kt * 4 + nt) * 32 + lane], acc[nt]);
    }
#pragma unroll
    for (int nt = 0; nt < 4; ++nt) {
      int colf = nt * 16 + r;
      float b1v = b1[colf];
#pragma unroll
      for (int v = 0; v < 8; ++v)
        lt[(v + 8 * hi) * HID + colf] = acc[nt][v] + b1v;
    }
    lds_fence();
    float sum = 0.f, sq = 0.f;
#pragma unroll 8
    for (int j = 0; j < 32; ++j) { float t = lt[r * HID + hi * 32 + j]; sum += t; sq += t * t; }
    sum += __shfl_xor(sum, 16);
    sq  += __shfl_xor(sq, 16);
    float mean = sum * (1.f / 64.f);
    float rs   = rsqrtf(sq * (1.f / 64.f) - mean * mean + EPSLN);
#pragma unroll 8
    for (int j = 0; j < 32; ++j) {
      int f = hi * 32 + j;
      float t = (lt[r * HID + f] - mean) * rs * gg[f] + bt[f];
      lh[r * HID + f] = (_Float16)fmaxf(t, 0.f);
    }
    lds_fence();
    Frag16 a2[2];
#pragma unroll
    for (int k2 = 0; k2 < 2; ++k2) {
      int f0 = k2 * 32 + hi * 8;
      a2[k2].u[0] = *(const uint4*)&lh[r * HID + f0];
      a2[k2].u[1] = *(const uint4*)&lh[r * HID + f0 + 16];
    }
    v8f acc2[4];
#pragma unroll
    for (int nt = 0; nt < 4; ++nt) acc2[nt] = vzero;
#pragma unroll
    for (int k2 = 0; k2 < 2; ++k2)
#pragma unroll
      for (int nt = 0; nt < 4; ++nt)
        acc2[nt] = wmma16(a2[k2].v, W2f[(k2 * 4 + nt) * 32 + lane], acc2[nt]);
#pragma unroll
    for (int nt = 0; nt < 4; ++nt) {
      int colf = nt * 16 + r;
      float b2v = b2[colf];
#pragma unroll
      for (int v = 0; v < 8; ++v)
        lt[(v + 8 * hi) * HID + colf] = acc2[nt][v] + b2v;
    }
    lds_fence();
    // + identity, final LayerNorm, write f32 output
    float s2 = 0.f, q2 = 0.f;
#pragma unroll 8
    for (int j = 0; j < 32; ++j) {
      int f = hi * 32 + j;
      float t = lt[r * HID + f] + x[(size_t)node * HID + f];
      lt[r * HID + f] = t;
      s2 += t; q2 += t * t;
    }
    lds_fence();
    s2 += __shfl_xor(s2, 16);
    q2 += __shfl_xor(q2, 16);
    float m2  = s2 * (1.f / 64.f);
    float rs2 = rsqrtf(q2 * (1.f / 64.f) - m2 * m2 + EPSLN);
#pragma unroll 8
    for (int j = 0; j < 32; ++j) {
      int f = hi * 32 + j;
      out[(size_t)node * HID + f] = (lt[r * HID + f] - m2) * rs2 * lng[f] + lnb[f];
    }
  }
}

// ------------------------------- host driver -------------------------------

extern "C" void kernel_launch(void* const* d_in, const int* in_sizes, int n_in,
                              void* d_out, int out_size, void* d_ws, size_t ws_size,
                              hipStream_t stream) {
  (void)n_in; (void)out_size; (void)ws_size;
  const float* x     = (const float*)d_in[0];
  const int*   ei    = (const int*)d_in[1];
  const float* ea    = (const float*)d_in[2];
  const int*   batch = (const int*)d_in[3];
  const int N  = in_sizes[0] / HID;
  const int E  = in_sizes[2] / 2;
  const int Nb = N / NB_G;

  // params: conv c key order within flattened pytree (sorted keys)
  auto CV = [&](int c, int k) { return (const float*)d_in[4 + 12 * c + k]; };
  // k: 0 bW,1 bb,2 W1,3 W2,4 b1,5 b2,6 bt,7 g,8 nb,9 ng,10 sW,11 sb
  const float* ln_b  = (const float*)d_in[40];
  const float* ln_g  = (const float*)d_in[41];
  MamP mp;
  mp.A_log = (const float*)d_in[42]; mp.D       = (const float*)d_in[43];
  mp.conv_W= (const float*)d_in[44]; mp.conv_b  = (const float*)d_in[45];
  mp.dt_W  = (const float*)d_in[46]; mp.dt_b    = (const float*)d_in[47];
  mp.fnorm_w=(const float*)d_in[48]; mp.in_W    = (const float*)d_in[49];
  mp.norm_w= (const float*)d_in[50]; mp.out_W   = (const float*)d_in[51];
  mp.xproj_W=(const float*)d_in[52];
  const float* mW1 = (const float*)d_in[53];
  const float* mW2 = (const float*)d_in[54];
  const float* mb1 = (const float*)d_in[55];
  const float* mb2 = (const float*)d_in[56];
  const float* mbt = (const float*)d_in[57];
  const float* mg  = (const float*)d_in[58];
  const float* theta = (const float*)d_in[59];

  // workspace carve (256B aligned)
  char* wp = (char*)d_ws;
  auto carve = [&](size_t bytes) -> char* {
    char* p = wp;
    wp += (bytes + 255) & ~size_t(255);
    return p;
  };
  _Float16* xh   = (_Float16*)carve((size_t)N * HID * 2);
  _Float16* w1h  = (_Float16*)carve((size_t)3 * 16 * 512 * 2);
  _Float16* w2h  = (_Float16*)carve((size_t)3 * 8 * 512 * 2);
  _Float16* mw1h = (_Float16*)carve((size_t)32 * 512 * 2);
  _Float16* mw2h = (_Float16*)carve((size_t)8 * 512 * 2);
  float*    agg  = (float*)carve((size_t)3 * N * HID * 4);
  float*    G    = (float*)carve((size_t)NB_G * HID * HID * 4);
  float*    aX   = (float*)carve((size_t)NB_G * KTOK * HID * 4);
  _Float16* xmh  = (_Float16*)carve((size_t)NB_G * HID * 2);
  _Float16* xcat = (_Float16*)carve((size_t)N * 256 * 2);

  EdgeP ep;
  NodeP np;
  for (int c = 0; c < 3; ++c) {
    ep.p[c][0] = CV(c, 4);  ep.p[c][1] = CV(c, 5);
    ep.p[c][2] = CV(c, 7);  ep.p[c][3] = CV(c, 6);
    ep.p[c][4] = CV(c, 10); ep.p[c][5] = CV(c, 11);
    ep.p[c][6] = CV(c, 0);  ep.p[c][7] = CV(c, 1);
    np.ng[c] = CV(c, 9);
    np.nb[c] = CV(c, 8);
  }

  // 1) f16 copies / weight fragment packing
  k_cvt_f16<<<dim3((N * HID + 255) / 256), dim3(256), 0, stream>>>(x, xh, N * HID);
  for (int c = 0; c < 3; ++c) {
    k_prep_frag<<<dim3(32), dim3(256), 0, stream>>>(CV(c, 2), w1h + c * 16 * 512, 128);
    k_prep_frag<<<dim3(16), dim3(256), 0, stream>>>(CV(c, 3), w2h + c * 8 * 512, 64);
  }
  k_prep_frag<<<dim3(64), dim3(256), 0, stream>>>(mW1, mw1h, 256);
  k_prep_frag<<<dim3(16), dim3(256), 0, stream>>>(mW2, mw2h, 64);
  // 2) zero aggregation buffers
  k_zero_f32<<<dim3(2048), dim3(256), 0, stream>>>(agg, 3 * N * HID);
  // 3) edge MLPs (WMMA) + scatter
  k_edge<<<dim3(2000), dim3(128), 0, stream>>>(xh, ei, ea, w1h, w2h, ep, agg,
                                               E / 16, E, N);
  // 4) per-node LN -> xcat columns 0..191
  k_node_ln<<<dim3(512), dim3(256), 0, stream>>>(x, agg, xcat, np, N);
  // 5) Gram matrices, alpha_X = theta^T (X^T X), Mamba, broadcast x_m
  k_gram<<<dim3(NB_G * 16), dim3(256), 0, stream>>>(x, G, Nb);
  k_alphax<<<dim3(NB_G), dim3(256), 0, stream>>>(theta, G, aX);
  k_mamba<<<dim3(NB_G), dim3(128), 0, stream>>>(aX, mp, xmh);
  k_fill_xm<<<dim3((N * HID + 255) / 256), dim3(256), 0, stream>>>(xmh, batch, xcat,
                                                                   N * HID);
  // 6) merge MLP (WMMA) + residual + final LN -> d_out (f32)
  k_merge<<<dim3(800), dim3(128), 0, stream>>>(xcat, mw1h, mw2h, mb1, mg, mbt, mb2,
                                               x, ln_g, ln_b, (float*)d_out, N / 16);
}
